// IncrementalAttentionBlock_89275190215314
// MI455X (gfx1250) — compile-verified, matrix operand order checked by reading, not run on hardware
//
#include <hip/hip_runtime.h>
#include <hip/hip_bf16.h>

typedef __attribute__((ext_vector_type(16))) _Float16 v16h;
typedef __attribute__((ext_vector_type(8)))  _Float16 v8h;
typedef __attribute__((ext_vector_type(8)))  float    v8f;
typedef __attribute__((ext_vector_type(4)))  unsigned int u32x4;
typedef __attribute__((ext_vector_type(4)))  int i32x4;
typedef __attribute__((ext_vector_type(8)))  int i32x8;

#define TT 4096
#define DM 1024
#define NH 16
#define HD 64
#define LWIN 64
#define LTR 16
#define NCH 125   // chunks 0..124 are ever visible

#define WMMA(a, b, c) __builtin_amdgcn_wmma_f32_16x16x32_f16( \
    false, (a), false, (b), (short)0, (c), false, false)

#if defined(__has_builtin)
#if __has_builtin(__builtin_amdgcn_tensor_load_to_lds)
#define HAVE_TDM 1
#endif
#endif
#if defined(HAVE_TDM) && __has_include(<hip/amd_detail/amd_gfx1250_TDM.h>)
#define TDM_ARGS6 1   // therock-10.0 headers => clang-23 6-arg builtin
#endif

// Load a 16-element f16 WMMA operand fragment for lane (lm, half):
// elements 0..7  = p[0..7]   -> K = base + 8*half + j
// elements 8..15 = p[16..23] -> K = base + 16 + 8*half + j
__device__ __forceinline__ v16h ldfrag(const _Float16* p) {
  v8h lo = *(const v8h*)p;
  v8h hi = *(const v8h*)(p + 16);
  v16h r;
#pragma unroll
  for (int j = 0; j < 8; ++j) { r[j] = lo[j]; r[j + 8] = hi[j]; }
  return r;
}

__device__ __forceinline__ v8h zero8() {
  v8h z;
#pragma unroll
  for (int j = 0; j < 8; ++j) z[j] = (_Float16)0.f;
  return z;
}

#ifdef HAVE_TDM
// TDM 2D tile load: rows x 64-f16 tile of a row-major [*, row_stride] f16 tensor
// into LDS at byte offset lds_off. Rows >= rows_avail are OOB -> zero-filled by TDM.
// D# packing per CDNA5 ISA (2D: groups 2/3 NULL->zero).
__device__ __forceinline__ void tdm_load_2d(unsigned lds_off, const _Float16* gptr,
                                            unsigned rows_avail, unsigned row_stride,
                                            unsigned tile_w, unsigned tile_h) {
  unsigned long long ga = (unsigned long long)(const void*)gptr;
  u32x4 g0;
  g0[0] = 1u;                                   // count=1, user mode, no gather
  g0[1] = lds_off;                              // lds_addr [63:32]
  g0[2] = (unsigned)(ga & 0xffffffffu);         // global_addr[31:0]
  g0[3] = (unsigned)((ga >> 32) & 0x01ffffffu)  // global_addr[56:32]
          | (2u << 30);                         // type=2 ("image")
  i32x8 g1;
  unsigned td0 = tile_w;        // tensor_dim0 (elements)
  unsigned td1 = rows_avail;    // tensor_dim1 (rows available before OOB-zero)
  g1[0] = (int)(1u << 16);                                  // data_size=1 (2 bytes)
  g1[1] = (int)((td0 & 0xffffu) << 16);                     // tensor_dim0[15:0] @ [31:16]
  g1[2] = (int)((td0 >> 16) | ((td1 & 0xffffu) << 16));     // dim0[31:16] | dim1[15:0]
  g1[3] = (int)((td1 >> 16) | ((tile_w & 0xffffu) << 16));  // dim1[31:16] | tile_dim0
  g1[4] = (int)(tile_h & 0xffffu);                          // tile_dim1; tile_dim2=0
  g1[5] = (int)row_stride;                                  // tensor_dim0_stride[31:0]
  g1[6] = 0;                                                // stride hi / dim1_stride lo
  g1[7] = 0;
  i32x4 z4;
#pragma unroll
  for (int j = 0; j < 4; ++j) z4[j] = 0;
#ifdef TDM_ARGS6
  i32x8 z8;
#pragma unroll
  for (int j = 0; j < 8; ++j) z8[j] = 0;
  __builtin_amdgcn_tensor_load_to_lds(g0, g1, z4, z4, z8, 0);
#else
  __builtin_amdgcn_tensor_load_to_lds(g0, g1, z4, z4, 0);
#endif
}
#endif

// ---------------------------------------------------------------- converts
__global__ void cvt_f32_f16(const float* __restrict__ s, _Float16* __restrict__ d, int n) {
  int i = blockIdx.x * blockDim.x + threadIdx.x;
  if (i < n) d[i] = (_Float16)s[i];
}

// ---------------------------------------------------------------- GEMM: C = A @ B^T + bias
// A: [M][K] f16 row-major, B: [N][K] f16 row-major. One wave -> 16x64 tile
// (A fragment reused across 4 N-tiles per k-step).
__global__ __launch_bounds__(32)
void gemm_f16out(const _Float16* __restrict__ A, const _Float16* __restrict__ B,
                 const float* __restrict__ bias, _Float16* __restrict__ C,
                 int N, int K) {
  const int lane = threadIdx.x, lm = lane & 15, half = lane >> 4;
  const int m0 = blockIdx.x * 16, n0 = blockIdx.y * 64;
  const _Float16* arow = A + (size_t)(m0 + lm) * K + half * 8;
  const _Float16* brow = B + (size_t)(n0 + lm) * K + half * 8;
  v8f acc[4];
#pragma unroll
  for (int nt = 0; nt < 4; ++nt)
#pragma unroll
    for (int r = 0; r < 8; ++r) acc[nt][r] = 0.f;
  for (int kk = 0; kk < K; kk += 32) {
    v16h a = ldfrag(arow + kk);
#pragma unroll
    for (int nt = 0; nt < 4; ++nt) {
      v16h b = ldfrag(brow + (size_t)nt * 16 * K + kk);
      acc[nt] = WMMA(a, b, acc[nt]);
    }
  }
#pragma unroll
  for (int nt = 0; nt < 4; ++nt) {
    const int n = n0 + nt * 16 + lm;
    const float bv = bias ? bias[n] : 0.f;
#pragma unroll
    for (int r = 0; r < 8; ++r) {
      int m = m0 + r + 8 * half;
      C[(size_t)m * N + n] = (_Float16)(acc[nt][r] + bv);
    }
  }
}

__global__ __launch_bounds__(32)
void gemm_f32out(const _Float16* __restrict__ A, const _Float16* __restrict__ B,
                 const float* __restrict__ bias, float* __restrict__ C,
                 int N, int K) {
  const int lane = threadIdx.x, lm = lane & 15, half = lane >> 4;
  const int m0 = blockIdx.x * 16, n0 = blockIdx.y * 64;
  const _Float16* arow = A + (size_t)(m0 + lm) * K + half * 8;
  const _Float16* brow = B + (size_t)(n0 + lm) * K + half * 8;
  v8f acc[4];
#pragma unroll
  for (int nt = 0; nt < 4; ++nt)
#pragma unroll
    for (int r = 0; r < 8; ++r) acc[nt][r] = 0.f;
  for (int kk = 0; kk < K; kk += 32) {
    v16h a = ldfrag(arow + kk);
#pragma unroll
    for (int nt = 0; nt < 4; ++nt) {
      v16h b = ldfrag(brow + (size_t)nt * 16 * K + kk);
      acc[nt] = WMMA(a, b, acc[nt]);
    }
  }
#pragma unroll
  for (int nt = 0; nt < 4; ++nt) {
    const int n = n0 + nt * 16 + lm;
    const float bv = bias ? bias[n] : 0.f;
#pragma unroll
    for (int r = 0; r < 8; ++r) {
      int m = m0 + r + 8 * half;
      C[(size_t)m * N + n] = acc[nt][r] + bv;
    }
  }
}

// ---------------------------------------------------------------- gate: sigmoid(x @ Wg^T + bg), f32
__global__ void gate_kernel(const float* __restrict__ x, const float* __restrict__ Wg,
                            const float* __restrict__ bg, float* __restrict__ G) {
  int idx = blockIdx.x * blockDim.x + threadIdx.x;  // TT*NH threads
  int t = idx >> 4, hh = idx & 15;
  const float* xr = x + (size_t)t * DM;
  const float* wr = Wg + (size_t)hh * DM;
  float a = bg[hh];
  for (int d = 0; d < DM; ++d) a += xr[d] * wr[d];
  G[idx] = 1.f / (1.f + __expf(-a));
}

// ---------------------------------------------------------------- chunk means (32-token K/V means)
__global__ void chunk_mean_kernel(const _Float16* __restrict__ Kh, const _Float16* __restrict__ Vh,
                                  _Float16* __restrict__ cmK, _Float16* __restrict__ cmV) {
  int j = blockIdx.x;  // 0..NCH-1
  for (int c = threadIdx.x; c < DM; c += blockDim.x) {
    float sk = 0.f, sv = 0.f;
    for (int i = 0; i < 32; ++i) {
      size_t t = (size_t)j * 32 + i;
      sk += (float)Kh[t * DM + c];
      sv += (float)Vh[t * DM + c];
    }
    cmK[(size_t)j * DM + c] = (_Float16)(sk * (1.f / 32.f));
    cmV[(size_t)j * DM + c] = (_Float16)(sv * (1.f / 32.f));
  }
}

// ---------------------------------------------------------------- fused local+remote attention
// One wave per (16-query tile, head). Local: sliding window [t-64, t-1].
// Remote: rc(t) = max(0, (t-96)/32 + 1), uniform within a 16-aligned tile.
// V tiles staged to LDS via the Tensor Data Mover (overlapped with QK^T WMMAs).
__global__ __launch_bounds__(32)
void attn_kernel(const _Float16* __restrict__ Qh, const _Float16* __restrict__ Kh,
                 const _Float16* __restrict__ Vh, const _Float16* __restrict__ cmK,
                 const _Float16* __restrict__ cmV, const float* __restrict__ G,
                 _Float16* __restrict__ Ao) {
  __shared__ __align__(16) _Float16 sP[16 * 96];   // local probs, padded to K=96
  __shared__ __align__(16) _Float16 sV[96 * 64];   // local V tile
  __shared__ __align__(16) _Float16 sPR[16 * 32];  // remote probs, padded to K=32
  __shared__ __align__(16) _Float16 sRV[32 * 64];  // remote V tile

  const int lane = threadIdx.x, lm = lane & 15, half = lane >> 4;
  const int t0 = blockIdx.x * 16, h = blockIdx.y;
  const float scale = 0.125f;  // 1/sqrt(64)

  const int kstart = (t0 >= LWIN) ? (t0 - LWIN) : 0;  // keys kstart..kstart+79 cover all windows
  const int rc = (t0 >= 96) ? (((t0 - 96) >> 5) + 1) : 0;
  const int cnt = (rc < LTR) ? rc : LTR;
  const int jlo = rc - cnt;

#ifdef HAVE_TDM
  // Kick off async V-tile DMA now; it overlaps the score WMMAs below.
  // Rows past (TT - kstart) are outside tensor_dim1 -> zero-filled.
  tdm_load_2d((unsigned)(size_t)&sV[0], Vh + (size_t)kstart * DM + h * HD,
              (unsigned)(TT - kstart), DM, HD, 96);
  if (rc > 0)
    tdm_load_2d((unsigned)(size_t)&sRV[0], cmV + (size_t)jlo * DM + h * HD,
                (unsigned)cnt, DM, HD, 32);
#endif

  const _Float16* qrow = Qh + (size_t)(t0 + lm) * DM + h * HD + half * 8;
  const v16h qa0 = ldfrag(qrow);
  const v16h qa1 = ldfrag(qrow + 32);

  // ---- local scores: S = Q K^T over 5 key blocks of 16
  v8f s[5];
#pragma unroll
  for (int b = 0; b < 5; ++b) {
    int key = kstart + b * 16 + lm;  // always < TT
    const _Float16* krow = Kh + (size_t)key * DM + h * HD + half * 8;
    v16h kb0 = ldfrag(krow);
    v16h kb1 = ldfrag(krow + 32);
    v8f z;
#pragma unroll
    for (int r = 0; r < 8; ++r) z[r] = 0.f;
    z = WMMA(qa0, kb0, z);
    z = WMMA(qa1, kb1, z);
    s[b] = z;
  }
  // mask: key in [t-64, t-1]
#pragma unroll
  for (int r = 0; r < 8; ++r) {
    int t = t0 + r + 8 * half;
#pragma unroll
    for (int b = 0; b < 5; ++b) {
      int kk = kstart + b * 16 + lm;
      float v = s[b][r] * scale;
      bool valid = (kk < t) && (kk + LWIN >= t);
      s[b][r] = valid ? v : -1e30f;
    }
  }
  // softmax per row (row spread across 16 lanes of a half-wave) -> P into LDS
#pragma unroll
  for (int r = 0; r < 8; ++r) {
    float mx = -1e30f;
#pragma unroll
    for (int b = 0; b < 5; ++b) mx = fmaxf(mx, s[b][r]);
    for (int msk = 1; msk <= 8; msk <<= 1) mx = fmaxf(mx, __shfl_xor(mx, msk, 32));
    float p[5], sum = 0.f;
#pragma unroll
    for (int b = 0; b < 5; ++b) { p[b] = __expf(s[b][r] - mx); sum += p[b]; }
    for (int msk = 1; msk <= 8; msk <<= 1) sum += __shfl_xor(sum, msk, 32);
    float inv = 1.f / sum;
    int m = r + 8 * half;
#pragma unroll
    for (int b = 0; b < 5; ++b) sP[m * 96 + b * 16 + lm] = (_Float16)(p[b] * inv);
  }
  for (int i = lane; i < 256; i += 32) sP[(i >> 4) * 96 + 80 + (i & 15)] = (_Float16)0.f;
#ifndef HAVE_TDM
  // fallback: manual V-tile staging [96][64]
  for (int i = lane; i < 768; i += 32) {
    int kk = i >> 3, d8 = (i & 7) * 8;
    int key = kstart + kk;
    v8h v = (key < TT) ? *(const v8h*)(Vh + (size_t)key * DM + h * HD + d8) : zero8();
    *(v8h*)(&sV[kk * 64 + d8]) = v;
  }
#endif
#ifdef HAVE_TDM
  __builtin_amdgcn_s_wait_tensorcnt(0);
#endif
  __syncthreads();

  // ---- local PV: O = P V  (K-dim 96 = 3 wmma steps, 4 d-tiles)
  v8f lo[4];
#pragma unroll
  for (int dt = 0; dt < 4; ++dt)
#pragma unroll
    for (int r = 0; r < 8; ++r) lo[dt][r] = 0.f;
#pragma unroll
  for (int ks = 0; ks < 3; ++ks) {
    v16h ap;
#pragma unroll
    for (int j = 0; j < 8; ++j) {
      int k0 = ks * 32 + half * 8 + j;
      ap[j]     = sP[lm * 96 + k0];
      ap[j + 8] = sP[lm * 96 + k0 + 16];
    }
#pragma unroll
    for (int dt = 0; dt < 4; ++dt) {
      int col = dt * 16 + lm;
      v16h bvf;
#pragma unroll
      for (int j = 0; j < 8; ++j) {
        int k0 = ks * 32 + half * 8 + j;
        bvf[j]     = sV[k0 * 64 + col];
        bvf[j + 8] = sV[(k0 + 16) * 64 + col];
      }
      lo[dt] = WMMA(ap, bvf, lo[dt]);
    }
  }

  // ---- remote (latent ring) attention; rc uniform across the tile
  v8f ro[4];
#pragma unroll
  for (int dt = 0; dt < 4; ++dt)
#pragma unroll
    for (int r = 0; r < 8; ++r) ro[dt][r] = 0.f;
  if (rc > 0) {
    const _Float16* crow = cmK + (size_t)(jlo + lm) * DM + h * HD + half * 8;
    v16h cb0 = ldfrag(crow);
    v16h cb1 = ldfrag(crow + 32);
    v8f sr;
#pragma unroll
    for (int r = 0; r < 8; ++r) sr[r] = 0.f;
    sr = WMMA(qa0, cb0, sr);
    sr = WMMA(qa1, cb1, sr);
    const bool slotok = lm < cnt;
#pragma unroll
    for (int r = 0; r < 8; ++r) {
      float v = slotok ? sr[r] * scale : -1e30f;
      float mx = v;
      for (int msk = 1; msk <= 8; msk <<= 1) mx = fmaxf(mx, __shfl_xor(mx, msk, 32));
      float e = __expf(v - mx);
      float sum = e;
      for (int msk = 1; msk <= 8; msk <<= 1) sum += __shfl_xor(sum, msk, 32);
      sPR[(r + 8 * half) * 32 + lm] = (_Float16)(e / sum);
    }
    for (int i = lane; i < 256; i += 32) sPR[(i >> 4) * 32 + 16 + (i & 15)] = (_Float16)0.f;
#ifndef HAVE_TDM
    for (int i = lane; i < 256; i += 32) {  // fallback: stage cmV tile [32][64]
      int kk = i >> 3, d8 = (i & 7) * 8;
      v8h v = (kk < cnt) ? *(const v8h*)(cmV + (size_t)(jlo + kk) * DM + h * HD + d8) : zero8();
      *(v8h*)(&sRV[kk * 64 + d8]) = v;
    }
#endif
    __syncthreads();
    v16h apr;
#pragma unroll
    for (int j = 0; j < 8; ++j) {
      int k0 = half * 8 + j;
      apr[j]     = sPR[lm * 32 + k0];
      apr[j + 8] = sPR[lm * 32 + k0 + 16];
    }
#pragma unroll
    for (int dt = 0; dt < 4; ++dt) {
      int col = dt * 16 + lm;
      v16h bvf;
#pragma unroll
      for (int j = 0; j < 8; ++j) {
        int k0 = half * 8 + j;
        bvf[j]     = sRV[k0 * 64 + col];
        bvf[j + 8] = sRV[(k0 + 16) * 64 + col];
      }
      ro[dt] = WMMA(apr, bvf, ro[dt]);
    }
  }

  // ---- gate-combine and store (t==0 row has no context -> exact 0)
#pragma unroll
  for (int r = 0; r < 8; ++r) {
    int m = r + 8 * half;
    int t = t0 + m;
    float g = G[t * NH + h];
#pragma unroll
    for (int dt = 0; dt < 4; ++dt) {
      float val = g * lo[dt][r] + (1.f - g) * ro[dt][r];
      if (t == 0) val = 0.f;
      Ao[(size_t)t * DM + h * HD + dt * 16 + lm] = (_Float16)val;
    }
  }
}

// ---------------------------------------------------------------- launch
extern "C" void kernel_launch(void* const* d_in, const int* in_sizes, int n_in,
                              void* d_out, int out_size, void* d_ws, size_t ws_size,
                              hipStream_t stream) {
  (void)in_sizes; (void)n_in; (void)out_size; (void)ws_size;
  const float* x  = (const float*)d_in[0];
  const float* Wq = (const float*)d_in[1];
  const float* bq = (const float*)d_in[2];
  const float* Wk = (const float*)d_in[3];
  const float* bk = (const float*)d_in[4];
  const float* Wv = (const float*)d_in[5];
  const float* bv = (const float*)d_in[6];
  const float* Wo = (const float*)d_in[7];
  const float* bo = (const float*)d_in[8];
  const float* Wg = (const float*)d_in[9];
  const float* bg = (const float*)d_in[10];
  float* out = (float*)d_out;

  char* p = (char*)d_ws;
  auto alloc = [&](size_t bytes) -> void* {
    void* r = (void*)p;
    p += (bytes + 255) & ~(size_t)255;
    return r;
  };
  _Float16* xh  = (_Float16*)alloc((size_t)TT * DM * 2);
  _Float16* Wqh = (_Float16*)alloc((size_t)DM * DM * 2);
  _Float16* Wkh = (_Float16*)alloc((size_t)DM * DM * 2);
  _Float16* Wvh = (_Float16*)alloc((size_t)DM * DM * 2);
  _Float16* Woh = (_Float16*)alloc((size_t)DM * DM * 2);
  _Float16* Qh  = (_Float16*)alloc((size_t)TT * DM * 2);
  _Float16* Kh  = (_Float16*)alloc((size_t)TT * DM * 2);
  _Float16* Vh  = (_Float16*)alloc((size_t)TT * DM * 2);
  float*    Gg  = (float*)   alloc((size_t)TT * NH * 4);
  _Float16* cmK = (_Float16*)alloc((size_t)128 * DM * 2);
  _Float16* cmV = (_Float16*)alloc((size_t)128 * DM * 2);
  _Float16* Ao  = (_Float16*)alloc((size_t)TT * DM * 2);

  // 1) f32 -> f16 converts
  cvt_f32_f16<<<(TT * DM + 255) / 256, 256, 0, stream>>>(x,  xh,  TT * DM);
  cvt_f32_f16<<<(DM * DM + 255) / 256, 256, 0, stream>>>(Wq, Wqh, DM * DM);
  cvt_f32_f16<<<(DM * DM + 255) / 256, 256, 0, stream>>>(Wk, Wkh, DM * DM);
  cvt_f32_f16<<<(DM * DM + 255) / 256, 256, 0, stream>>>(Wv, Wvh, DM * DM);
  cvt_f32_f16<<<(DM * DM + 255) / 256, 256, 0, stream>>>(Wo, Woh, DM * DM);

  // 2) Q/K/V projections via WMMA (16x64 tile per wave)
  dim3 gg(TT / 16, DM / 64);
  gemm_f16out<<<gg, 32, 0, stream>>>(xh, Wqh, bq, Qh, DM, DM);
  gemm_f16out<<<gg, 32, 0, stream>>>(xh, Wkh, bk, Kh, DM, DM);
  gemm_f16out<<<gg, 32, 0, stream>>>(xh, Wvh, bv, Vh, DM, DM);

  // 3) gates (f32 for precision; tiny)
  gate_kernel<<<(TT * NH) / 256, 256, 0, stream>>>(x, Wg, bg, Gg);

  // 4) remote chunk means
  chunk_mean_kernel<<<NCH, 256, 0, stream>>>(Kh, Vh, cmK, cmV);

  // 5) fused local+remote attention (TDM-staged V tiles)
  attn_kernel<<<dim3(TT / 16, NH), 32, 0, stream>>>(Qh, Kh, Vh, cmK, cmV, Gg, Ao);

  // 6) output projection (f32 out)
  gemm_f32out<<<gg, 32, 0, stream>>>(Ao, Woh, bo, out, DM, DM);
}
